// SelfAttention_75118978007109
// MI455X (gfx1250) — matrix operand order repaired and checked by reading, not compiled, not run on hardware
//
#include <hip/hip_runtime.h>

// ---------------------------------------------------------------------------
// MI455X (gfx1250) self-attention forward.
// All GEMMs run on v_wmma_f32_16x16x32_bf16 (wave32, f32 accumulate).
// GEMM wave tile 64x64 (4x4 WMMA tiles) -> 1.0 b128 loads per WMMA.
// ---------------------------------------------------------------------------

typedef __bf16 bf16_t;
typedef __attribute__((ext_vector_type(16))) __bf16 v16bf;
typedef __attribute__((ext_vector_type(8)))  __bf16 v8bf;
typedef __attribute__((ext_vector_type(8)))  float  v8f;

#define S_LEN  2048
#define NHEADS 16
#define HDIM   128
#define HID    2048
#define BATCH  2
#define MROWS  4096      // B*S
#define N_QKV  6144      // 3*H

static __device__ __forceinline__ v16bf cat8(v8bf lo, v8bf hi) {
  v16bf r;
#pragma unroll
  for (int i = 0; i < 8; ++i) { r[i] = lo[i]; r[i + 8] = hi[i]; }
  return r;
}

static __device__ __forceinline__ v8f wmma_bf16(v16bf a, v16bf b, v8f c) {
  // D = A(16x32 bf16) * B(32x16 bf16) + C(16x16 f32)
  return __builtin_amdgcn_wmma_f32_16x16x32_bf16(false, a, false, b,
                                                 (short)0, c, false, false);
}

// ------------------------------- fp32 -> bf16 ------------------------------
__global__ void cvt_kernel(const float* __restrict__ src, bf16_t* __restrict__ dst, int n) {
  int i = blockIdx.x * 256 + threadIdx.x;
  if (i < n) dst[i] = (bf16_t)src[i];
}

// ------------------------- RoPE + query scaling ----------------------------
// q,k layout: [B*NH, S, D].  d in [0,64): pair (d, d+64).
__global__ void rope_kernel(bf16_t* __restrict__ q, bf16_t* __restrict__ k,
                            const float* __restrict__ cosd, const float* __restrict__ sind) {
  int idx = blockIdx.x * 256 + threadIdx.x;          // B*NH*S*64 threads
  if (idx >= BATCH * NHEADS * S_LEN * (HDIM / 2)) return;
  int d  = idx & 63;
  int s  = (idx >> 6) & (S_LEN - 1);
  int bh = idx >> 17;
  float c  = cosd[s * 64 + d];
  float sn = sind[s * 64 + d];
  size_t base = ((size_t)bh * S_LEN + s) * HDIM;
  const float scale = 0.08838834764831845f;          // 1/sqrt(128)
  float q1 = (float)q[base + d], q2 = (float)q[base + 64 + d];
  q[base + d]      = (bf16_t)((q1 * c - q2 * sn) * scale);
  q[base + 64 + d] = (bf16_t)((q2 * c + q1 * sn) * scale);
  float k1 = (float)k[base + d], k2 = (float)k[base + 64 + d];
  k[base + d]      = (bf16_t)(k1 * c - k2 * sn);
  k[base + 64 + d] = (bf16_t)(k2 * c + k1 * sn);
}

// ------------------------------ GEMM (bf16) --------------------------------
// out[m,n] = sum_k A[m,k] * Bw[n,k]   (both K-major, K % 32 == 0)
// WG = 256 thr = 8 waves arranged 2(M) x 4(N); wave tile 64x64; WG tile 128x256.
// EPI==0: plain f32 store. EPI==1: scatter into Q/K ([B,NH,S,D]) and V^T ([B,NH,D,S]).
template <int EPI>
__global__ __launch_bounds__(256)
void gemm_kernel(const bf16_t* __restrict__ A, const bf16_t* __restrict__ Bw,
                 int N, int K, float* __restrict__ outF,
                 bf16_t* __restrict__ qb, bf16_t* __restrict__ kb, bf16_t* __restrict__ vtb) {
  const int lane = threadIdx.x & 31;
  const int wid  = threadIdx.x >> 5;
  const int lq = lane & 15, hf = lane >> 4;
  const int mBase = blockIdx.y * 128 + (wid & 1) * 64;
  const int nBase = blockIdx.x * 256 + (wid >> 1) * 64;

  v8f acc[4][4];
#pragma unroll
  for (int i = 0; i < 4; ++i)
#pragma unroll
    for (int j = 0; j < 4; ++j) acc[i][j] = (v8f){};

  for (int k0 = 0; k0 < K; k0 += 32) {
    v16bf af[4], bw[4];
#pragma unroll
    for (int tm = 0; tm < 4; ++tm) {
      const bf16_t* p = A + (size_t)(mBase + tm * 16 + lq) * K + (k0 + hf * 8);
      af[tm] = cat8(*(const v8bf*)p, *(const v8bf*)(p + 16));   // K[0..7],K[16..23] per half
    }
#pragma unroll
    for (int tn = 0; tn < 4; ++tn) {
      const bf16_t* p = Bw + (size_t)(nBase + tn * 16 + lq) * K + (k0 + hf * 8);
      bw[tn] = cat8(*(const v8bf*)p, *(const v8bf*)(p + 16));   // K[0..7],K[16..23] per half
    }
#pragma unroll
    for (int tm = 0; tm < 4; ++tm)
#pragma unroll
      for (int tn = 0; tn < 4; ++tn)
        acc[tm][tn] = wmma_bf16(af[tm], bw[tn], acc[tm][tn]);
  }

#pragma unroll
  for (int tm = 0; tm < 4; ++tm)
#pragma unroll
    for (int tn = 0; tn < 4; ++tn)
#pragma unroll
      for (int j = 0; j < 8; ++j) {
        int m = mBase + tm * 16 + j + 8 * hf;     // C layout: row = elem + 8*half
        int n = nBase + tn * 16 + lq;             // col = lane&15
        float val = acc[tm][tn][j];
        if (EPI == 0) {
          outF[(size_t)m * N + n] = val;
        } else {
          int b = m >> 11, s = m & (S_LEN - 1);
          int which = n >> 11, r = n & 2047;
          int h = r >> 7, d = r & 127;
          size_t bh = (size_t)(b * NHEADS + h);
          if (which == 0)      qb[(bh * S_LEN + s) * HDIM + d]  = (bf16_t)val;
          else if (which == 1) kb[(bh * S_LEN + s) * HDIM + d]  = (bf16_t)val;
          else                 vtb[(bh * HDIM + d) * S_LEN + s] = (bf16_t)val;  // V transposed
        }
      }
}

// --------------------------- Flash attention -------------------------------
// grid = (S/128, B*NH). 8 waves; each wave owns 16 query rows, streams 64-key
// blocks: QK^T (16 WMMA) -> online softmax -> P via LDS relayout -> PV (16 WMMA).
__global__ __launch_bounds__(256)
void attn_kernel(const bf16_t* __restrict__ qb, const bf16_t* __restrict__ kb,
                 const bf16_t* __restrict__ vtb, bf16_t* __restrict__ ao) {
  __shared__ bf16_t pl[8 * 16 * 64];               // 16 KB: per-wave 16x64 P tile
  const int lane = threadIdx.x & 31;
  const int wid  = threadIdx.x >> 5;
  const int lq = lane & 15, hf = lane >> 4;
  const int bh = blockIdx.y;
  const int b = bh >> 4, h = bh & 15;
  const int q0 = blockIdx.x * 128 + wid * 16;

  // Preload Q A-fragments (16 rows x 128 head-dim = 4 chained frags)
  v16bf aq[4];
  {
    const bf16_t* qrow = qb + ((size_t)bh * S_LEN + q0 + lq) * HDIM;
#pragma unroll
    for (int c = 0; c < 4; ++c) {
      const bf16_t* p = qrow + c * 32 + hf * 8;
      aq[c] = cat8(*(const v8bf*)p, *(const v8bf*)(p + 16));
    }
  }

  v8f acc[8];
#pragma unroll
  for (int t = 0; t < 8; ++t) acc[t] = (v8f){};
  float mrow[8], lrow[8];
#pragma unroll
  for (int j = 0; j < 8; ++j) { mrow[j] = -__builtin_inff(); lrow[j] = 0.0f; }

  const int kend = (blockIdx.x + 1) * 128;         // block-causal bound (uniform in WG)
  for (int key0 = 0; key0 < kend; key0 += 64) {
    // --- scores S = Q K^T : 16x64, f32 ---
    v8f sc[4];
#pragma unroll
    for (int t = 0; t < 4; ++t) sc[t] = (v8f){};
#pragma unroll
    for (int c = 0; c < 4; ++c)
#pragma unroll
      for (int t = 0; t < 4; ++t) {
        const bf16_t* p = kb + ((size_t)bh * S_LEN + key0 + t * 16 + lq) * HDIM + c * 32 + hf * 8;
        v16bf bk = cat8(*(const v8bf*)p, *(const v8bf*)(p + 16));
        sc[t] = wmma_bf16(aq[c], bk, sc[t]);
      }
    // --- causal mask (key > query -> -inf) ---
#pragma unroll
    for (int t = 0; t < 4; ++t) {
      int keyn = key0 + t * 16 + lq;
#pragma unroll
      for (int j = 0; j < 8; ++j)
        if (keyn > q0 + j + 8 * hf) sc[t][j] = -__builtin_inff();
    }
    // --- online softmax: row max across 4 tiles + 16 lanes of each half ---
    float mt[8];
#pragma unroll
    for (int j = 0; j < 8; ++j)
      mt[j] = fmaxf(fmaxf(sc[0][j], sc[1][j]), fmaxf(sc[2][j], sc[3][j]));
#pragma unroll
    for (int off = 1; off < 16; off <<= 1)
#pragma unroll
      for (int j = 0; j < 8; ++j)
        mt[j] = fmaxf(mt[j], __shfl_xor(mt[j], off, 32));
    float corr[8];
#pragma unroll
    for (int j = 0; j < 8; ++j) {
      float mn = fmaxf(mrow[j], mt[j]);
      corr[j] = __expf(mrow[j] - mn);
      mrow[j] = mn;
    }
    float rs[8];
#pragma unroll
    for (int j = 0; j < 8; ++j) rs[j] = 0.0f;
#pragma unroll
    for (int t = 0; t < 4; ++t)
#pragma unroll
      for (int j = 0; j < 8; ++j) {
        float p = __expf(sc[t][j] - mrow[j]);
        sc[t][j] = p; rs[j] += p;
      }
#pragma unroll
    for (int off = 1; off < 16; off <<= 1)
#pragma unroll
      for (int j = 0; j < 8; ++j)
        rs[j] += __shfl_xor(rs[j], off, 32);
#pragma unroll
    for (int j = 0; j < 8; ++j) lrow[j] = lrow[j] * corr[j] + rs[j];
#pragma unroll
    for (int t = 0; t < 8; ++t)
#pragma unroll
      for (int j = 0; j < 8; ++j) acc[t][j] *= corr[j];

    // --- P: C-fragment layout -> LDS -> A-fragment layout ---
    __syncthreads();
#pragma unroll
    for (int t = 0; t < 4; ++t)
#pragma unroll
      for (int j = 0; j < 8; ++j)
        pl[wid * 1024 + (j + 8 * hf) * 64 + t * 16 + lq] = (bf16_t)sc[t][j];
    __syncthreads();

    // --- O += P V : K-dim = 64 keys (2 chained frags), N = 128 head-dim ---
#pragma unroll
    for (int c = 0; c < 2; ++c) {
      const bf16_t* pp = &pl[wid * 1024 + lq * 64 + c * 32 + hf * 8];
      v16bf ap = cat8(*(const v8bf*)pp, *(const v8bf*)(pp + 16));
#pragma unroll
      for (int tn = 0; tn < 8; ++tn) {
        const bf16_t* pv = vtb + ((size_t)bh * HDIM + tn * 16 + lq) * S_LEN + key0 + c * 32 + hf * 8;
        v16bf bv = cat8(*(const v8bf*)pv, *(const v8bf*)(pv + 16));
        acc[tn] = wmma_bf16(ap, bv, acc[tn]);
      }
    }
  }

  // --- normalize and store to [B, S, H] (bf16, feeds output projection) ---
  float inv[8];
#pragma unroll
  for (int j = 0; j < 8; ++j) inv[j] = 1.0f / lrow[j];
#pragma unroll
  for (int tn = 0; tn < 8; ++tn)
#pragma unroll
    for (int j = 0; j < 8; ++j) {
      int row = q0 + j + 8 * hf;
      ao[((size_t)b * S_LEN + row) * HID + h * HDIM + tn * 16 + lq] = (bf16_t)(acc[tn][j] * inv[j]);
    }
}

// ------------------------------- launcher ----------------------------------
extern "C" void kernel_launch(void* const* d_in, const int* in_sizes, int n_in,
                              void* d_out, int out_size, void* d_ws, size_t ws_size,
                              hipStream_t stream) {
  (void)in_sizes; (void)n_in; (void)out_size; (void)ws_size;
  const float* x    = (const float*)d_in[0];
  const float* qkvw = (const float*)d_in[1];
  const float* ow   = (const float*)d_in[2];
  const float* cosd = (const float*)d_in[3];
  const float* sind = (const float*)d_in[4];
  // d_in[5] = causal_mask: causality computed analytically.

  const size_t n_x  = (size_t)MROWS * HID;                 // 8.39 M
  const size_t n_w  = (size_t)N_QKV * HID;                 // 12.58 M
  const size_t n_ow = (size_t)HID * HID;                   // 4.19 M
  const size_t n_h  = (size_t)BATCH * NHEADS * S_LEN * HDIM; // 8.39 M

  bf16_t* x_bf  = (bf16_t*)d_ws;
  bf16_t* w_bf  = x_bf + n_x;
  bf16_t* ow_bf = w_bf + n_w;
  bf16_t* qb    = ow_bf + n_ow;
  bf16_t* kb    = qb + n_h;
  bf16_t* vt    = kb + n_h;
  bf16_t* ao    = vt + n_h;            // total ~117 MB of workspace

  cvt_kernel<<<(int)((n_x  + 255) / 256), 256, 0, stream>>>(x,    x_bf,  (int)n_x);
  cvt_kernel<<<(int)((n_w  + 255) / 256), 256, 0, stream>>>(qkvw, w_bf,  (int)n_w);
  cvt_kernel<<<(int)((n_ow + 255) / 256), 256, 0, stream>>>(ow,   ow_bf, (int)n_ow);

  // QKV projection, scatter into Q/K ([B,NH,S,D]) and V^T ([B,NH,D,S])
  gemm_kernel<1><<<dim3(N_QKV / 256, MROWS / 128), 256, 0, stream>>>(
      x_bf, w_bf, N_QKV, HID, nullptr, qb, kb, vt);

  const int nrope = BATCH * NHEADS * S_LEN * (HDIM / 2);
  rope_kernel<<<(nrope + 255) / 256, 256, 0, stream>>>(qb, kb, cosd, sind);

  attn_kernel<<<dim3(S_LEN / 128, BATCH * NHEADS), 256, 0, stream>>>(qb, kb, vt, ao);

  // Output projection -> fp32 d_out
  gemm_kernel<0><<<dim3(HID / 256, MROWS / 128), 256, 0, stream>>>(
      ao, ow_bf, HID, HID, (float*)d_out, nullptr, nullptr, nullptr);
}
